// SpikeMixtralDecoderLayer_5171140624986
// MI455X (gfx1250) — matrix operand order repaired and checked by reading, not compiled, not run on hardware
//
#include <hip/hip_runtime.h>
#include <hip/hip_bf16.h>
#include <stdint.h>

#define S_LEN 2048
#define HID   2048
#define NHQ   32
#define HD    64
#define NKV   8
#define NEXP  8
#define FFN   4096
#define EPSV  1e-5f

typedef __bf16 bf16x16 __attribute__((ext_vector_type(16)));
typedef float  floatx8 __attribute__((ext_vector_type(8)));

union Frag { uint4 q[2]; bf16x16 v; };

#if defined(__HIP_DEVICE_COMPILE__) && __has_builtin(__builtin_amdgcn_sched_barrier)
#define SCHED_FENCE() __builtin_amdgcn_sched_barrier(0)
#else
#define SCHED_FENCE()
#endif

__device__ inline unsigned short f2bf(float f) {
  __bf16 h = (__bf16)f;  // native v_cvt (RNE)
  return __builtin_bit_cast(unsigned short, h);
}
__device__ inline float bf2f(unsigned short u) {
  return (float)__builtin_bit_cast(__bf16, u);
}
__device__ inline floatx8 vzero8() {
  floatx8 v;
#pragma unroll
  for (int i = 0; i < 8; ++i) v[i] = 0.f;
  return v;
}

// LDS byte offset of a shared-memory pointer (LDS aperture: addr[31:0] == LDS offset)
__device__ inline unsigned lds_off(const void* p) { return (unsigned)(uintptr_t)p; }

// CDNA5 async global->LDS copy, 16B per lane, tracked by ASYNCcnt (§15.18 op 98)
__device__ inline void async_b128(unsigned dst_lds, const void* src) {
  asm volatile("global_load_async_to_lds_b128 %0, %1, off"
               :: "v"(dst_lds), "v"(src)
               : "memory");
}
template<int N>
__device__ inline void s_wait_async() {
  asm volatile("s_wait_asynccnt %0" :: "n"(N) : "memory");
}

// ---------------- RMSNorm (f32 in) -> bf16 out ----------------
__global__ void rmsnorm_to_bf16(const float* __restrict__ x, const float* __restrict__ w,
                                unsigned short* __restrict__ out) {
  __shared__ float red[256];
  int row = blockIdx.x, tid = threadIdx.x;
  const float* xr = x + (size_t)row * HID;
  float s = 0.f;
#pragma unroll
  for (int i = 0; i < HID / 256; ++i) { float v = xr[tid + i * 256]; s += v * v; }
  red[tid] = s;
  __syncthreads();
  for (int st = 128; st > 0; st >>= 1) {
    if (tid < st) red[tid] += red[tid + st];
    __syncthreads();
  }
  float scale = rsqrtf(red[0] / (float)HID + EPSV);
  unsigned short* orow = out + (size_t)row * HID;
#pragma unroll
  for (int i = 0; i < HID / 256; ++i) {
    int c = tid + i * 256;
    orow[c] = f2bf(xr[c] * scale * w[c]);
  }
}

// ---------------- WMMA GEMM: C[M,N] = A_bf16[M,K] * W_f32[N,K]^T ----------------
// BM=128, BN=64, BK=32; double-buffered LDS; A staged via async-to-LDS;
// W prefetched into registers one K-step ahead. M,N multiples of 128/64.
__global__ __launch_bounds__(256)
void gemm_wmma(const unsigned short* __restrict__ A, const float* __restrict__ W,
               float* __restrict__ Cf, unsigned short* __restrict__ Cb,
               const float* __restrict__ resid, int M, int N, int K) {
  __shared__ unsigned short As[2][128 * 40];
  __shared__ unsigned short Ws[2][64 * 40];
  int tid = threadIdx.x, wave = tid >> 5, lane = tid & 31, lr = lane & 15, hi = lane >> 4;
  int bm = blockIdx.y * 128, bn = blockIdx.x * 64;

  int ar = tid >> 1, ah = tid & 1;
  const unsigned short* aptr = A + (size_t)(bm + ar) * K + ah * 16;
  unsigned a_lds[2] = { lds_off(&As[0][ar * 40 + ah * 16]),
                        lds_off(&As[1][ar * 40 + ah * 16]) };
  int wn = tid >> 2, wq = tid & 3;
  const float* wptr = W + (size_t)(bn + wn) * K + wq * 8;

  floatx8 acc[4];
#pragma unroll
  for (int i = 0; i < 4; ++i) acc[i] = vzero8();

  // prologue: stage tile 0
  async_b128(a_lds[0], aptr);
  async_b128(a_lds[0] + 16, aptr + 8);
  { alignas(16) unsigned short t[8];
#pragma unroll
    for (int j = 0; j < 8; ++j) t[j] = f2bf(wptr[j]);
    *(uint4*)&Ws[0][wn * 40 + wq * 8] = *(const uint4*)t; }

  int nsteps = K / 32;
  for (int s = 0; s < nsteps; ++s) {
    int cur = s & 1, nxt = cur ^ 1;
    bool have_next = (s + 1 < nsteps);
    s_wait_async<0>();
    __syncthreads();
    alignas(16) unsigned short wt[8];
    if (have_next) {
      int kt = (s + 1) * 32;
      async_b128(a_lds[nxt], aptr + kt);
      async_b128(a_lds[nxt] + 16, aptr + kt + 8);
      const float* wp = wptr + kt;
#pragma unroll
      for (int j = 0; j < 8; ++j) wt[j] = f2bf(wp[j]);
    }
    Frag af;
    { const unsigned short* ap = &As[cur][(wave * 16 + lr) * 40 + hi * 8];
      af.q[0] = *(const uint4*)ap; af.q[1] = *(const uint4*)(ap + 16); }
    Frag bf[4];
#pragma unroll
    for (int nt = 0; nt < 4; ++nt) {
      const unsigned short* bp = &Ws[cur][(nt * 16 + lr) * 40 + hi * 8];
      bf[nt].q[0] = *(const uint4*)bp; bf[nt].q[1] = *(const uint4*)(bp + 16);
    }
    SCHED_FENCE();
#pragma unroll
    for (int nt = 0; nt < 4; ++nt)
      acc[nt] = __builtin_amdgcn_wmma_f32_16x16x32_bf16(false, af.v, false, bf[nt].v,
                                                        (short)0, acc[nt], false, false);
    if (have_next) *(uint4*)&Ws[nxt][wn * 40 + wq * 8] = *(const uint4*)wt;
  }
#pragma unroll
  for (int nt = 0; nt < 4; ++nt) {
#pragma unroll
    for (int r = 0; r < 8; ++r) {
      int grow = bm + wave * 16 + r + hi * 8, gcol = bn + nt * 16 + lr;
      float v = acc[nt][r];
      if (Cf) {
        if (resid) v += resid[(size_t)grow * N + gcol];
        Cf[(size_t)grow * N + gcol] = v;
      }
      if (Cb) Cb[(size_t)grow * N + gcol] = f2bf(v);
    }
  }
}

// ---------------- MoE up: H = silu(X*W1^T) * (X*W3^T), gathered rows ----------------
__global__ __launch_bounds__(256)
void gemm_dual_silu(const unsigned short* __restrict__ X, const float* __restrict__ W1,
                    const float* __restrict__ W3, unsigned short* __restrict__ Hout,
                    const int* __restrict__ cntp, const int* __restrict__ map,
                    int N, int K) {
  int cnt = cntp[0];
  int bm = blockIdx.y * 128;
  if (bm >= cnt) return;
  __shared__ unsigned short As[2][128 * 40];
  __shared__ unsigned short W1s[2][64 * 40];
  __shared__ unsigned short W3s[2][64 * 40];
  int tid = threadIdx.x, wave = tid >> 5, lane = tid & 31, lr = lane & 15, hi = lane >> 4;
  int bn = blockIdx.x * 64;

  int ar = tid >> 1, ah = tid & 1, gr = bm + ar;
  int tok = (gr < cnt) ? map[gr] : map[0];  // clamp; masked at store
  const unsigned short* aptr = X + (size_t)tok * K + ah * 16;
  unsigned a_lds[2] = { lds_off(&As[0][ar * 40 + ah * 16]),
                        lds_off(&As[1][ar * 40 + ah * 16]) };
  int wn = tid >> 2, wq = tid & 3;
  const float* w1p = W1 + (size_t)(bn + wn) * K + wq * 8;
  const float* w3p = W3 + (size_t)(bn + wn) * K + wq * 8;

  floatx8 a1[4], a3[4];
#pragma unroll
  for (int i = 0; i < 4; ++i) { a1[i] = vzero8(); a3[i] = vzero8(); }

  async_b128(a_lds[0], aptr);
  async_b128(a_lds[0] + 16, aptr + 8);
  { alignas(16) unsigned short t1[8], t3[8];
#pragma unroll
    for (int j = 0; j < 8; ++j) { t1[j] = f2bf(w1p[j]); t3[j] = f2bf(w3p[j]); }
    *(uint4*)&W1s[0][wn * 40 + wq * 8] = *(const uint4*)t1;
    *(uint4*)&W3s[0][wn * 40 + wq * 8] = *(const uint4*)t3; }

  int nsteps = K / 32;
  for (int s = 0; s < nsteps; ++s) {
    int cur = s & 1, nxt = cur ^ 1;
    bool have_next = (s + 1 < nsteps);
    s_wait_async<0>();
    __syncthreads();
    alignas(16) unsigned short t1[8], t3[8];
    if (have_next) {
      int kt = (s + 1) * 32;
      async_b128(a_lds[nxt], aptr + kt);
      async_b128(a_lds[nxt] + 16, aptr + kt + 8);
#pragma unroll
      for (int j = 0; j < 8; ++j) { t1[j] = f2bf(w1p[kt + j]); t3[j] = f2bf(w3p[kt + j]); }
    }
    Frag af;
    { const unsigned short* ap = &As[cur][(wave * 16 + lr) * 40 + hi * 8];
      af.q[0] = *(const uint4*)ap; af.q[1] = *(const uint4*)(ap + 16); }
#pragma unroll
    for (int nt = 0; nt < 4; ++nt) {
      Frag b1, b3;
      const unsigned short* bp1 = &W1s[cur][(nt * 16 + lr) * 40 + hi * 8];
      const unsigned short* bp3 = &W3s[cur][(nt * 16 + lr) * 40 + hi * 8];
      b1.q[0] = *(const uint4*)bp1; b1.q[1] = *(const uint4*)(bp1 + 16);
      b3.q[0] = *(const uint4*)bp3; b3.q[1] = *(const uint4*)(bp3 + 16);
      a1[nt] = __builtin_amdgcn_wmma_f32_16x16x32_bf16(false, af.v, false, b1.v,
                                                       (short)0, a1[nt], false, false);
      a3[nt] = __builtin_amdgcn_wmma_f32_16x16x32_bf16(false, af.v, false, b3.v,
                                                       (short)0, a3[nt], false, false);
    }
    if (have_next) {
      *(uint4*)&W1s[nxt][wn * 40 + wq * 8] = *(const uint4*)t1;
      *(uint4*)&W3s[nxt][wn * 40 + wq * 8] = *(const uint4*)t3;
    }
  }
#pragma unroll
  for (int nt = 0; nt < 4; ++nt) {
#pragma unroll
    for (int r = 0; r < 8; ++r) {
      int grow = bm + wave * 16 + r + hi * 8, gcol = bn + nt * 16 + lr;
      if (grow < cnt) {
        float x = a1[nt][r];
        float h = x * (1.f / (1.f + __expf(-x))) * a3[nt][r];
        Hout[(size_t)grow * N + gcol] = f2bf(h);
      }
    }
  }
}

// ---------------- MoE down: out[tok] += wts * (H * W2^T), scattered rows ----------------
__global__ __launch_bounds__(256)
void gemm_scatter(const unsigned short* __restrict__ Hin, const float* __restrict__ W2,
                  float* __restrict__ Out, const int* __restrict__ cntp,
                  const int* __restrict__ map, const float* __restrict__ wts,
                  int N, int K) {
  int cnt = cntp[0];
  int bm = blockIdx.y * 128;
  if (bm >= cnt) return;
  __shared__ unsigned short As[2][128 * 40];
  __shared__ unsigned short Ws[2][64 * 40];
  int tid = threadIdx.x, wave = tid >> 5, lane = tid & 31, lr = lane & 15, hi = lane >> 4;
  int bn = blockIdx.x * 64;

  int ar = tid >> 1, ah = tid & 1, gr = bm + ar;
  int arow = (gr < cnt) ? gr : 0;  // clamp; masked at store
  const unsigned short* aptr = Hin + (size_t)arow * K + ah * 16;
  unsigned a_lds[2] = { lds_off(&As[0][ar * 40 + ah * 16]),
                        lds_off(&As[1][ar * 40 + ah * 16]) };
  int wn = tid >> 2, wq = tid & 3;
  const float* wptr = W2 + (size_t)(bn + wn) * K + wq * 8;

  floatx8 acc[4];
#pragma unroll
  for (int i = 0; i < 4; ++i) acc[i] = vzero8();

  async_b128(a_lds[0], aptr);
  async_b128(a_lds[0] + 16, aptr + 8);
  { alignas(16) unsigned short t[8];
#pragma unroll
    for (int j = 0; j < 8; ++j) t[j] = f2bf(wptr[j]);
    *(uint4*)&Ws[0][wn * 40 + wq * 8] = *(const uint4*)t; }

  int nsteps = K / 32;
  for (int s = 0; s < nsteps; ++s) {
    int cur = s & 1, nxt = cur ^ 1;
    bool have_next = (s + 1 < nsteps);
    s_wait_async<0>();
    __syncthreads();
    alignas(16) unsigned short wt[8];
    if (have_next) {
      int kt = (s + 1) * 32;
      async_b128(a_lds[nxt], aptr + kt);
      async_b128(a_lds[nxt] + 16, aptr + kt + 8);
#pragma unroll
      for (int j = 0; j < 8; ++j) wt[j] = f2bf(wptr[kt + j]);
    }
    Frag af;
    { const unsigned short* ap = &As[cur][(wave * 16 + lr) * 40 + hi * 8];
      af.q[0] = *(const uint4*)ap; af.q[1] = *(const uint4*)(ap + 16); }
    Frag bf[4];
#pragma unroll
    for (int nt = 0; nt < 4; ++nt) {
      const unsigned short* bp = &Ws[cur][(nt * 16 + lr) * 40 + hi * 8];
      bf[nt].q[0] = *(const uint4*)bp; bf[nt].q[1] = *(const uint4*)(bp + 16);
    }
    SCHED_FENCE();
#pragma unroll
    for (int nt = 0; nt < 4; ++nt)
      acc[nt] = __builtin_amdgcn_wmma_f32_16x16x32_bf16(false, af.v, false, bf[nt].v,
                                                        (short)0, acc[nt], false, false);
    if (have_next) *(uint4*)&Ws[nxt][wn * 40 + wq * 8] = *(const uint4*)wt;
  }
#pragma unroll
  for (int nt = 0; nt < 4; ++nt) {
#pragma unroll
    for (int r = 0; r < 8; ++r) {
      int grow = bm + wave * 16 + r + hi * 8, gcol = bn + nt * 16 + lr;
      if (grow < cnt) {
        int tok = map[grow];
        size_t o = (size_t)tok * N + gcol;
        Out[o] += wts[grow] * acc[nt][r];
      }
    }
  }
}

// ---------------- RoPE (f32 in -> bf16 out) ----------------
__global__ void rope_to_bf16(const float* __restrict__ src, unsigned short* __restrict__ dst,
                             int nheads) {
  int idx = blockIdx.x * blockDim.x + threadIdx.x;
  int total = S_LEN * nheads * 32;
  if (idx >= total) return;
  int i = idx & 31;
  int t = idx >> 5;
  int h = t % nheads;
  int s = t / nheads;
  size_t base = ((size_t)s * nheads + h) * HD;
  float x1 = src[base + i], x2 = src[base + i + 32];
  float inv = __powf(1.0e6f, -((float)i) / 32.0f);
  float ang = (float)s * inv;
  float c, sn;
  __sincosf(ang, &c, &sn);
  dst[base + i]      = f2bf(x1 * c - x2 * sn);
  dst[base + i + 32] = f2bf(x2 * c + x1 * sn);
}

__global__ void f32_to_bf16(const float* __restrict__ src, unsigned short* __restrict__ dst,
                            int n) {
  int i = blockIdx.x * blockDim.x + threadIdx.x;
  if (i < n) dst[i] = f2bf(src[i]);
}

// ---------------- Flash attention (causal, GQA h -> h % NKV) ----------------
__global__ __launch_bounds__(128)
void flash_attn(const unsigned short* __restrict__ Q, const unsigned short* __restrict__ Kc,
                const unsigned short* __restrict__ Vc, unsigned short* __restrict__ O) {
  __shared__ unsigned short Qs[64 * 72];  // [qrow][dim]
  __shared__ unsigned short Ks[32 * 72];  // [key][dim]
  __shared__ unsigned short Vs[64 * 40];  // transposed [dim][key]
  __shared__ unsigned short Ps[64 * 40];  // [qrow][key]
  int head = blockIdx.x, qblk = blockIdx.y, qb0 = qblk * 64;
  int kvh = head & (NKV - 1);
  int tid = threadIdx.x, wave = tid >> 5, lane = tid & 31, lr = lane & 15, hi = lane >> 4;
  { // async Q tile (stays resident)
    int r = tid >> 1, h = tid & 1;
    const unsigned short* p = Q + (size_t)(qb0 + r) * (NHQ * HD) + head * HD + h * 32;
    unsigned d = lds_off(&Qs[r * 72 + h * 32]);
    async_b128(d, p);
    async_b128(d + 16, p + 8);
    async_b128(d + 32, p + 16);
    async_b128(d + 48, p + 24);
  }
  float m[8], l[8];
#pragma unroll
  for (int r = 0; r < 8; ++r) { m[r] = -1e30f; l[r] = 0.f; }
  floatx8 o[4];
#pragma unroll
  for (int i = 0; i < 4; ++i) o[i] = vzero8();
  int kbend = (qb0 + 63) >> 5;
  for (int kb = 0; kb <= kbend; ++kb) {
    __syncthreads();  // prior reads of Ks/Vs done before overwrite
    int key = tid >> 2, q4 = tid & 3;
    { // async K tile
      const unsigned short* p = Kc + (size_t)(kb * 32 + key) * (NKV * HD) + kvh * HD + q4 * 16;
      unsigned d = lds_off(&Ks[key * 72 + q4 * 16]);
      async_b128(d, p);
      async_b128(d + 16, p + 8);
    }
    { // V tile, transposed through registers
      const unsigned short* p = Vc + (size_t)(kb * 32 + key) * (NKV * HD) + kvh * HD + q4 * 16;
#pragma unroll
      for (int j = 0; j < 16; ++j) Vs[(q4 * 16 + j) * 40 + key] = p[j];
    }
    s_wait_async<0>();
    __syncthreads();
    floatx8 sc[2];
    sc[0] = vzero8(); sc[1] = vzero8();
    Frag aq[2];
#pragma unroll
    for (int ksi = 0; ksi < 2; ++ksi) {
      const unsigned short* ap = &Qs[(wave * 16 + lr) * 72 + ksi * 32 + hi * 8];
      aq[ksi].q[0] = *(const uint4*)ap; aq[ksi].q[1] = *(const uint4*)(ap + 16);
    }
    Frag bk[4];
#pragma unroll
    for (int nt = 0; nt < 2; ++nt)
#pragma unroll
      for (int ksi = 0; ksi < 2; ++ksi) {
        const unsigned short* bp = &Ks[(nt * 16 + lr) * 72 + ksi * 32 + hi * 8];
        bk[nt * 2 + ksi].q[0] = *(const uint4*)bp;
        bk[nt * 2 + ksi].q[1] = *(const uint4*)(bp + 16);
      }
    SCHED_FENCE();
#pragma unroll
    for (int nt = 0; nt < 2; ++nt)
#pragma unroll
      for (int ksi = 0; ksi < 2; ++ksi)
        sc[nt] = __builtin_amdgcn_wmma_f32_16x16x32_bf16(false, aq[ksi].v, false,
                                                         bk[nt * 2 + ksi].v,
                                                         (short)0, sc[nt], false, false);
#pragma unroll
    for (int r = 0; r < 8; ++r) {
      int rowg = qb0 + wave * 16 + r + hi * 8;
      float s0 = sc[0][r] * 0.125f, s1 = sc[1][r] * 0.125f;
      int c0 = kb * 32 + lr, c1 = kb * 32 + 16 + lr;
      if (c0 > rowg) s0 = -1e30f;
      if (c1 > rowg) s1 = -1e30f;
      float mx = fmaxf(s0, s1);
#pragma unroll
      for (int off = 8; off > 0; off >>= 1) mx = fmaxf(mx, __shfl_xor(mx, off, 16));
      float mn = fmaxf(m[r], mx);
      float esc = __expf(m[r] - mn);
      float p0 = __expf(s0 - mn), p1 = __expf(s1 - mn);
      float rs = p0 + p1;
#pragma unroll
      for (int off = 8; off > 0; off >>= 1) rs += __shfl_xor(rs, off, 16);
      l[r] = l[r] * esc + rs;
      m[r] = mn;
#pragma unroll
      for (int nt = 0; nt < 4; ++nt) o[nt][r] *= esc;
      int rl = wave * 16 + r + hi * 8;
      Ps[rl * 40 + lr]      = f2bf(p0);
      Ps[rl * 40 + 16 + lr] = f2bf(p1);
    }
    __syncthreads();
    Frag pf;
    { const unsigned short* pp = &Ps[(wave * 16 + lr) * 40 + hi * 8];
      pf.q[0] = *(const uint4*)pp; pf.q[1] = *(const uint4*)(pp + 16); }
    Frag vf[4];
#pragma unroll
    for (int nt = 0; nt < 4; ++nt) {
      const unsigned short* vp = &Vs[(nt * 16 + lr) * 40 + hi * 8];
      vf[nt].q[0] = *(const uint4*)vp; vf[nt].q[1] = *(const uint4*)(vp + 16);
    }
    SCHED_FENCE();
#pragma unroll
    for (int nt = 0; nt < 4; ++nt)
      o[nt] = __builtin_amdgcn_wmma_f32_16x16x32_bf16(false, pf.v, false, vf[nt].v,
                                                      (short)0, o[nt], false, false);
  }
#pragma unroll
  for (int r = 0; r < 8; ++r) {
    float il = 1.f / l[r];
    int rowg = qb0 + wave * 16 + r + hi * 8;
#pragma unroll
    for (int nt = 0; nt < 4; ++nt)
      O[(size_t)rowg * (NHQ * HD) + head * HD + nt * 16 + lr] = f2bf(o[nt][r] * il);
  }
}

// ---------------- Router: softmax(x @ gate^T), top-2, build expert lists ----------------
__global__ void zero_counts(int* counts) {
  if (threadIdx.x < NEXP) counts[threadIdx.x] = 0;
}

__global__ void router_topk(const unsigned short* __restrict__ xn, const float* __restrict__ gw,
                            int* __restrict__ counts, int* __restrict__ lists,
                            float* __restrict__ wts) {
  int row = blockIdx.x, lane = threadIdx.x;
  float acc[NEXP];
#pragma unroll
  for (int e = 0; e < NEXP; ++e) acc[e] = 0.f;
  const unsigned short* xr = xn + (size_t)row * HID;
  for (int h = lane; h < HID; h += 32) {
    float xv = bf2f(xr[h]);
#pragma unroll
    for (int e = 0; e < NEXP; ++e) acc[e] += xv * gw[e * HID + h];
  }
#pragma unroll
  for (int e = 0; e < NEXP; ++e)
    for (int off = 16; off > 0; off >>= 1) acc[e] += __shfl_xor(acc[e], off, 32);
  if (lane == 0) {
    float mx = acc[0];
    for (int e = 1; e < NEXP; ++e) mx = fmaxf(mx, acc[e]);
    float p[NEXP];
    for (int e = 0; e < NEXP; ++e) p[e] = __expf(acc[e] - mx);
    int e0 = 0;
    for (int e = 1; e < NEXP; ++e) if (p[e] > p[e0]) e0 = e;
    int e1 = (e0 == 0) ? 1 : 0;
    for (int e = 0; e < NEXP; ++e) if (e != e0 && p[e] > p[e1]) e1 = e;
    float t = p[e0] + p[e1];
    int pos = atomicAdd(&counts[e0], 1);
    lists[e0 * S_LEN + pos] = row; wts[e0 * S_LEN + pos] = p[e0] / t;
    pos = atomicAdd(&counts[e1], 1);
    lists[e1 * S_LEN + pos] = row; wts[e1 * S_LEN + pos] = p[e1] / t;
  }
}

// ---------------- Launch ----------------
extern "C" void kernel_launch(void* const* d_in, const int* in_sizes, int n_in,
                              void* d_out, int out_size, void* d_ws, size_t ws_size,
                              hipStream_t stream) {
  (void)in_sizes; (void)n_in; (void)out_size; (void)ws_size;
  const float* hs  = (const float*)d_in[0];
  const float* ln1 = (const float*)d_in[1];
  const float* ln2 = (const float*)d_in[2];
  const float* wq  = (const float*)d_in[3];
  const float* wk  = (const float*)d_in[4];
  const float* wv  = (const float*)d_in[5];
  const float* wo  = (const float*)d_in[6];
  const float* gw  = (const float*)d_in[7];
  const float* w1  = (const float*)d_in[8];
  const float* w2  = (const float*)d_in[9];
  const float* w3  = (const float*)d_in[10];
  float* out = (float*)d_out;

  char* p = (char*)d_ws;
  auto alloc = [&](size_t bytes) -> char* {
    char* r = p;
    p += (bytes + 255) & ~(size_t)255;
    return r;
  };
  unsigned short* xnb  = (unsigned short*)alloc((size_t)S_LEN * HID * 2);
  float* qf            = (float*)alloc((size_t)S_LEN * NHQ * HD * 4);
  float* kf            = (float*)alloc((size_t)S_LEN * NKV * HD * 4);
  float* vf            = (float*)alloc((size_t)S_LEN * NKV * HD * 4);
  unsigned short* qb   = (unsigned short*)alloc((size_t)S_LEN * NHQ * HD * 2);
  unsigned short* kb   = (unsigned short*)alloc((size_t)S_LEN * NKV * HD * 2);
  unsigned short* vb   = (unsigned short*)alloc((size_t)S_LEN * NKV * HD * 2);
  unsigned short* ab   = (unsigned short*)alloc((size_t)S_LEN * NHQ * HD * 2);
  unsigned short* xnb2 = (unsigned short*)alloc((size_t)S_LEN * HID * 2);
  int* counts          = (int*)alloc(NEXP * 4);
  int* lists           = (int*)alloc((size_t)NEXP * S_LEN * 4);
  float* wtsb          = (float*)alloc((size_t)NEXP * S_LEN * 4);
  unsigned short* hbuf = (unsigned short*)alloc((size_t)S_LEN * FFN * 2);

  // attention block
  rmsnorm_to_bf16<<<S_LEN, 256, 0, stream>>>(hs, ln1, xnb);
  gemm_wmma<<<dim3(NHQ * HD / 64, S_LEN / 128), 256, 0, stream>>>(
      xnb, wq, qf, nullptr, nullptr, S_LEN, NHQ * HD, HID);
  gemm_wmma<<<dim3(NKV * HD / 64, S_LEN / 128), 256, 0, stream>>>(
      xnb, wk, kf, nullptr, nullptr, S_LEN, NKV * HD, HID);
  gemm_wmma<<<dim3(NKV * HD / 64, S_LEN / 128), 256, 0, stream>>>(
      xnb, wv, vf, nullptr, nullptr, S_LEN, NKV * HD, HID);
  rope_to_bf16<<<(S_LEN * NHQ * 32 + 255) / 256, 256, 0, stream>>>(qf, qb, NHQ);
  rope_to_bf16<<<(S_LEN * NKV * 32 + 255) / 256, 256, 0, stream>>>(kf, kb, NKV);
  f32_to_bf16<<<(S_LEN * NKV * HD + 255) / 256, 256, 0, stream>>>(vf, vb, S_LEN * NKV * HD);
  flash_attn<<<dim3(NHQ, S_LEN / 64), 128, 0, stream>>>(qb, kb, vb, ab);
  gemm_wmma<<<dim3(HID / 64, S_LEN / 128), 256, 0, stream>>>(
      ab, wo, out, nullptr, hs, S_LEN, HID, NHQ * HD);  // out = residual + attn@wo^T

  // MoE block
  rmsnorm_to_bf16<<<S_LEN, 256, 0, stream>>>(out, ln2, xnb2);
  zero_counts<<<1, 32, 0, stream>>>(counts);
  router_topk<<<S_LEN, 32, 0, stream>>>(xnb2, gw, counts, lists, wtsb);
  for (int e = 0; e < NEXP; ++e) {
    gemm_dual_silu<<<dim3(FFN / 64, S_LEN / 128), 256, 0, stream>>>(
        xnb2, w1 + (size_t)e * FFN * HID, w3 + (size_t)e * FFN * HID, hbuf,
        counts + e, lists + e * S_LEN, FFN, HID);
    gemm_scatter<<<dim3(HID / 64, S_LEN / 128), 256, 0, stream>>>(
        hbuf, w2 + (size_t)e * HID * FFN, out, counts + e, lists + e * S_LEN,
        wtsb + e * S_LEN, HID, FFN);
  }
}
